// MatAtt_68255620268340
// MI455X (gfx1250) — compile-verified, matrix operand order checked by reading, not run on hardware
//
#include <hip/hip_runtime.h>

#define DD 128
#define NB 32
#define LC 4096
#define LA 256

typedef __attribute__((ext_vector_type(16))) __bf16        v16bf;
typedef __attribute__((ext_vector_type(2)))  __bf16        bf2;
typedef __attribute__((ext_vector_type(8)))  float         v8f;
typedef __attribute__((ext_vector_type(4)))  float         f4v;
typedef __attribute__((ext_vector_type(4)))  unsigned int  u4v;
typedef __attribute__((ext_vector_type(8)))  unsigned int  u8v;

// ---------------------------------------------------------------------------
// WMMA fragment helpers (CDNA5 wave32, V_WMMA_F32_16X16X32_BF16, ISA 7.12.2)
// A/B 16-bit fragment: VGPR v packs K-pair; lanes 0-15 (h=0) hold K 0..7 &
// 16..23, lanes 16-31 (h=1) hold K 8..15 & 24..31.  In memory (K contiguous)
// that is two 16-byte runs per lane: k in [8h, 8h+8) and [16+8h, 16+8h+8).
// ---------------------------------------------------------------------------

// Row-major bf16 source, K contiguous: two b128 loads, no repacking.
__device__ __forceinline__ v16bf frag_row_bf(const __bf16* p, int h) {
    u4v a = *(const u4v*)(p + 8 * h);
    u4v b = *(const u4v*)(p + 16 + 8 * h);
    u8v u;
#pragma unroll
    for (int i = 0; i < 4; ++i) { u[i] = a[i]; u[4 + i] = b[i]; }
    return __builtin_bit_cast(v16bf, u);
}

// Row-major f32 source: two pairs of b128 loads + cvt_pk to bf16.
__device__ __forceinline__ v16bf frag_row_f32(const float* p, int h) {
    f4v a0 = ((const f4v*)(p + 8 * h))[0];
    f4v a1 = ((const f4v*)(p + 8 * h))[1];
    f4v b0 = ((const f4v*)(p + 16 + 8 * h))[0];
    f4v b1 = ((const f4v*)(p + 16 + 8 * h))[1];
    v16bf f;
#pragma unroll
    for (int i = 0; i < 4; ++i) {
        f[i]      = (__bf16)a0[i];
        f[4 + i]  = (__bf16)a1[i];
        f[8 + i]  = (__bf16)b0[i];
        f[12 + i] = (__bf16)b1[i];
    }
    return f;
}

// K-pair-packed LDS tile: lds2[p*S2 + c] holds (bf16 X[2p][c], bf16 X[2p+1][c]).
// Column fragment = 8 ds_load_b32, bit-cast straight into operand VGPRs.
__device__ __forceinline__ v16bf frag_col_pk(const unsigned int* lds2, int col,
                                             int h, int S2) {
    u8v u;
#pragma unroll
    for (int vp = 0; vp < 8; ++vp) {
        int p = (vp & 3) + h * 4 + (vp >> 2) * 8;   // K-pair index
        u[vp] = lds2[p * S2 + col];
    }
    return __builtin_bit_cast(v16bf, u);
}

__device__ __forceinline__ v8f wmma_bf16(v16bf a, v16bf b, v8f c) {
    return __builtin_amdgcn_wmma_f32_16x16x32_bf16(false, a, false, b, (short)0, c, false, false);
}

__device__ __forceinline__ void zero8(v8f& v) {
#pragma unroll
    for (int q = 0; q < 8; ++q) v[q] = 0.0f;
}

// ---------------------------------------------------------------------------
// K0: weight conversion (bf16; transposed where the B-operand needs
// row-contiguous columns)
// ---------------------------------------------------------------------------
__global__ void k_prep_weights(const float* __restrict__ Wct, const float* __restrict__ Wtc,
                               const float* __restrict__ w1c, const float* __restrict__ w1a,
                               __bf16* __restrict__ Wct_b, __bf16* __restrict__ Wtc_b,
                               __bf16* __restrict__ W1cT_b, __bf16* __restrict__ W1aT_b) {
    int i = blockIdx.x * blockDim.x + threadIdx.x;   // 0 .. 16383
    int r = i >> 7, c = i & 127;
    Wct_b[i] = (__bf16)Wct[i];
    Wtc_b[i] = (__bf16)Wtc[i];
    W1cT_b[c * DD + r] = (__bf16)w1c[i];
    W1aT_b[c * DD + r] = (__bf16)w1a[i];
}

// ---------------------------------------------------------------------------
// K1: Gram matrix  G = X^T X   ([L,128] -> [128,128], bf16 out, f32 accum)
// One workgroup/batch.  32x128 f32 chunk is loaded coalesced (b128), packed
// into K-pair uints with v_cvt_pk_bf16_f32, staged in LDS (stride 132 uints
// => half-wave groups hit disjoint bank sets), consumed as b32 column reads.
// ---------------------------------------------------------------------------
__global__ void __launch_bounds__(256) k_gram(const float* __restrict__ X, int L,
                                              __bf16* __restrict__ Gb) {
    const int S2 = 132;
    __shared__ unsigned int lds2[16 * S2];           // 8448 B
    const int b    = blockIdx.x;
    const float* Xb = X + (size_t)b * L * DD;
    const int tid  = threadIdx.x;
    const int wave = tid >> 5;
    const int lane = tid & 31;
    const int h    = lane >> 4;
    const int r    = lane & 15;

    v8f acc[8];
#pragma unroll
    for (int t = 0; t < 8; ++t) zero8(acc[t]);

    const int pr = tid >> 4;          // row-pair 0..15
    const int cb = (tid & 15) * 8;    // column base

    for (int kc = 0; kc < L / 32; ++kc) {
        const float* r0 = Xb + ((size_t)kc * 32 + 2 * pr) * DD + cb;
        const float* r1 = r0 + DD;
        f4v a0 = ((const f4v*)r0)[0], a1 = ((const f4v*)r0)[1];
        f4v b0 = ((const f4v*)r1)[0], b1 = ((const f4v*)r1)[1];
        u4v p0, p1;
#pragma unroll
        for (int i = 0; i < 4; ++i) {
            bf2 w0; w0[0] = (__bf16)a0[i]; w0[1] = (__bf16)b0[i];   // (even k, odd k)
            bf2 w1; w1[0] = (__bf16)a1[i]; w1[1] = (__bf16)b1[i];
            p0[i] = __builtin_bit_cast(unsigned int, w0);
            p1[i] = __builtin_bit_cast(unsigned int, w1);
        }
        unsigned int* dst = lds2 + pr * S2 + cb;
        ((u4v*)dst)[0] = p0;
        ((u4v*)dst)[1] = p1;
        __syncthreads();

        v16bf af = frag_col_pk(lds2, 16 * wave + r, h, S2);   // A = X^T (col m)
#pragma unroll
        for (int t = 0; t < 8; ++t) {
            v16bf bf = frag_col_pk(lds2, 16 * t + r, h, S2);  // B = X (col n)
            acc[t] = wmma_bf16(af, bf, acc[t]);
        }
        __syncthreads();
    }

    __bf16* G = Gb + (size_t)b * DD * DD;
#pragma unroll
    for (int t = 0; t < 8; ++t)
#pragma unroll
        for (int j = 0; j < 8; ++j)
            G[(16 * wave + j + h * 8) * DD + 16 * t + r] = (__bf16)acc[t][j];
}

// ---------------------------------------------------------------------------
// K2: Mt[b] = G[b] @ W^T   (128x128x128; G symmetric so Mt = (W @ G)^T)
// ---------------------------------------------------------------------------
__global__ void __launch_bounds__(256) k_mixmat(const __bf16* __restrict__ Gb,
                                                const __bf16* __restrict__ Wb,
                                                __bf16* __restrict__ Mt) {
    const int b    = blockIdx.x;
    const __bf16* G = Gb + (size_t)b * DD * DD;
    const int tid  = threadIdx.x;
    const int wave = tid >> 5;
    const int lane = tid & 31;
    const int h    = lane >> 4;
    const int r    = lane & 15;

    v8f acc[8];
#pragma unroll
    for (int t = 0; t < 8; ++t) zero8(acc[t]);

    for (int kc = 0; kc < 4; ++kc) {
        v16bf af = frag_row_bf(G + (16 * wave + r) * DD + kc * 32, h);
#pragma unroll
        for (int t = 0; t < 8; ++t) {
            // B[k][n] = W[n][k] -> row n of W, k-contiguous
            v16bf bf = frag_row_bf(Wb + (16 * t + r) * DD + kc * 32, h);
            acc[t] = wmma_bf16(af, bf, acc[t]);
        }
    }

    __bf16* M = Mt + (size_t)b * DD * DD;
#pragma unroll
    for (int t = 0; t < 8; ++t)
#pragma unroll
        for (int j = 0; j < 8; ++j)
            M[(16 * wave + j + h * 8) * DD + 16 * t + r] = (__bf16)acc[t][j];
}

// ---------------------------------------------------------------------------
// K3: h[b] = X[b] @ M[b]   ([L,128] @ [128,128] -> f32)
// grid = (L/128, B); A converted f32->bf16 on load, B from Mt rows.
// ---------------------------------------------------------------------------
__global__ void __launch_bounds__(256) k_hgemm(const float* __restrict__ X,
                                               const __bf16* __restrict__ Mt,
                                               float* __restrict__ H, int L) {
    const int b    = blockIdx.y;
    const float* Xb = X + (size_t)b * L * DD;
    const __bf16* Mb = Mt + (size_t)b * DD * DD;
    float* Hb = H + (size_t)b * L * DD;

    const int tid  = threadIdx.x;
    const int wave = tid >> 5;
    const int lane = tid & 31;
    const int h    = lane >> 4;
    const int r    = lane & 15;
    const int rowBase = blockIdx.x * 128 + 16 * wave;

    v8f acc[8];
#pragma unroll
    for (int t = 0; t < 8; ++t) zero8(acc[t]);

    for (int kc = 0; kc < 4; ++kc) {
        const float* arow = Xb + (size_t)(rowBase + r) * DD + kc * 32;
        if (kc < 3) __builtin_prefetch(arow + 32, 0, 3);   // global_prefetch_b8
        v16bf af = frag_row_f32(arow, h);
#pragma unroll
        for (int t = 0; t < 8; ++t) {
            // B[k][n] = M[k][n] = Mt[n][k] -> row n of Mt, k-contiguous
            v16bf bf = frag_row_bf(Mb + (16 * t + r) * DD + kc * 32, h);
            acc[t] = wmma_bf16(af, bf, acc[t]);
        }
    }

#pragma unroll
    for (int t = 0; t < 8; ++t)
#pragma unroll
        for (int j = 0; j < 8; ++j)
            Hb[(size_t)(rowBase + j + h * 8) * DD + 16 * t + r] = acc[t][j];
}

// ---------------------------------------------------------------------------
// K4: per-row score  t[m] = relu( relu(H@W1 + b1) . w2 + b2 )
// WMMA GEMM fused with the w2 matvec; shfl_xor masks 1..8 stay inside the
// 16-lane halves of the C/D layout (M = j + 8h, N = lane&15).
// ---------------------------------------------------------------------------
__global__ void __launch_bounds__(256) k_score(const float* __restrict__ H,
                                               const __bf16* __restrict__ W1T,
                                               const float* __restrict__ b1,
                                               const float* __restrict__ w2,
                                               const float* __restrict__ b2,
                                               float* __restrict__ T, int L) {
    const int b    = blockIdx.y;
    const float* Hb = H + (size_t)b * L * DD;
    const int tid  = threadIdx.x;
    const int wave = tid >> 5;
    const int lane = tid & 31;
    const int h    = lane >> 4;
    const int r    = lane & 15;
    const int rowBase = blockIdx.x * 128 + 16 * wave;

    v8f acc[8];
#pragma unroll
    for (int t = 0; t < 8; ++t) zero8(acc[t]);

    for (int kc = 0; kc < 4; ++kc) {
        v16bf af = frag_row_f32(Hb + (size_t)(rowBase + r) * DD + kc * 32, h);
#pragma unroll
        for (int t = 0; t < 8; ++t) {
            v16bf bf = frag_row_bf(W1T + (16 * t + r) * DD + kc * 32, h);
            acc[t] = wmma_bf16(af, bf, acc[t]);
        }
    }

    float rowpart[8];
#pragma unroll
    for (int j = 0; j < 8; ++j) rowpart[j] = 0.0f;
#pragma unroll
    for (int t = 0; t < 8; ++t) {
        float b1n = b1[16 * t + r];
        float w2n = w2[16 * t + r];
#pragma unroll
        for (int j = 0; j < 8; ++j) {
            float u = acc[t][j] + b1n;
            u = u > 0.0f ? u : 0.0f;           // relu(H@W1+b1)
            rowpart[j] += u * w2n;
        }
    }
    float red[8];
#pragma unroll
    for (int j = 0; j < 8; ++j) {
        float v = rowpart[j];
#pragma unroll
        for (int m = 1; m < 16; m <<= 1) v += __shfl_xor(v, m, 32);
        red[j] = v;
    }
    if (r == 0) {
        float bias2 = b2[0];
#pragma unroll
        for (int j = 0; j < 8; ++j) {
            float s = red[j] + bias2;
            T[(size_t)b * L + rowBase + j + h * 8] = s > 0.0f ? s : 0.0f;
        }
    }
}

// ---------------------------------------------------------------------------
// K5: softmax over length + weighted pooling  vec[d] = sum_l p[l]*H[l][d]
// ---------------------------------------------------------------------------
__global__ void __launch_bounds__(256) k_softpool(const float* __restrict__ T,
                                                  const float* __restrict__ H,
                                                  float* __restrict__ vecOut, int L) {
    __shared__ float red[256];
    const int b   = blockIdx.x;
    const int tid = threadIdx.x;
    const float* Tb = T + (size_t)b * L;
    const float* Hb = H + (size_t)b * L * DD;

    float mx = -3.4e38f;
    for (int l = tid; l < L; l += 256) mx = fmaxf(mx, Tb[l]);
    red[tid] = mx; __syncthreads();
    for (int s = 128; s > 0; s >>= 1) {
        if (tid < s) red[tid] = fmaxf(red[tid], red[tid + s]);
        __syncthreads();
    }
    float gmax = red[0]; __syncthreads();

    float se = 0.0f;
    for (int l = tid; l < L; l += 256) se += expf(Tb[l] - gmax);
    red[tid] = se; __syncthreads();
    for (int s = 128; s > 0; s >>= 1) {
        if (tid < s) red[tid] += red[tid + s];
        __syncthreads();
    }
    float gsum = red[0]; __syncthreads();

    const int d = tid & 127, part = tid >> 7;
    float a = 0.0f;
    for (int l = part; l < L; l += 2)
        a += expf(Tb[l] - gmax) * Hb[(size_t)l * DD + d];
    red[tid] = a; __syncthreads();
    if (part == 0) vecOut[(size_t)b * DD + d] = (red[tid] + red[tid + 128]) / gsum;
}

// ---------------------------------------------------------------------------
// K6: second attention.  Folds (H @ W) @ vec^T -> H @ (W @ vec):
//   v = W @ vecIn; att[l] = relu(H[l].v); p = softmax(att); rep = sum p*H
// ---------------------------------------------------------------------------
__global__ void __launch_bounds__(256) k_att2(const float* __restrict__ H,
                                              const float* __restrict__ Wmat,
                                              const float* __restrict__ vecIn,
                                              float* __restrict__ repOut, int L) {
    __shared__ float vsh[DD];
    __shared__ float att[LC];       // L <= LC
    __shared__ float red[256];
    const int b   = blockIdx.x;
    const int tid = threadIdx.x;
    const float* Hb = H + (size_t)b * L * DD;
    const float* vb = vecIn + (size_t)b * DD;

    if (tid < DD) {
        float s = 0.0f;
        for (int j = 0; j < DD; ++j) s += Wmat[tid * DD + j] * vb[j];
        vsh[tid] = s;
    }
    __syncthreads();

    for (int l = tid; l < L; l += 256) {
        const float* row = Hb + (size_t)l * DD;
        float s = 0.0f;
        for (int j = 0; j < DD; ++j) s += row[j] * vsh[j];
        att[l] = s > 0.0f ? s : 0.0f;
    }
    __syncthreads();

    float mx = -3.4e38f;
    for (int l = tid; l < L; l += 256) mx = fmaxf(mx, att[l]);
    red[tid] = mx; __syncthreads();
    for (int s = 128; s > 0; s >>= 1) {
        if (tid < s) red[tid] = fmaxf(red[tid], red[tid + s]);
        __syncthreads();
    }
    float gmax = red[0]; __syncthreads();

    float se = 0.0f;
    for (int l = tid; l < L; l += 256) se += expf(att[l] - gmax);
    red[tid] = se; __syncthreads();
    for (int s = 128; s > 0; s >>= 1) {
        if (tid < s) red[tid] += red[tid + s];
        __syncthreads();
    }
    float gsum = red[0]; __syncthreads();

    const int d = tid & 127, part = tid >> 7;
    float a = 0.0f;
    for (int l = part; l < L; l += 2)
        a += expf(att[l] - gmax) * Hb[(size_t)l * DD + d];
    red[tid] = a; __syncthreads();
    if (part == 0) repOut[(size_t)b * DD + d] = (red[tid] + red[tid + 128]) / gsum;
}

// ---------------------------------------------------------------------------
// K7: final = relu(concat(aspect_rep, context_rep) @ Wf + bf)   [32,256]
// ---------------------------------------------------------------------------
__global__ void k_final(const float* __restrict__ arep, const float* __restrict__ crep,
                        const float* __restrict__ Wf, const float* __restrict__ bf,
                        float* __restrict__ out) {
    const int b = blockIdx.x, n = threadIdx.x;   // n in 0..255
    float s = bf[n];
    const float* ar = arep + (size_t)b * DD;
    const float* cr = crep + (size_t)b * DD;
    for (int k = 0; k < DD; ++k) s += ar[k] * Wf[k * 256 + n];
    for (int k = 0; k < DD; ++k) s += cr[k] * Wf[(DD + k) * 256 + n];
    out[(size_t)b * 256 + n] = s > 0.0f ? s : 0.0f;
}

// ---------------------------------------------------------------------------
extern "C" void kernel_launch(void* const* d_in, const int* in_sizes, int n_in,
                              void* d_out, int out_size, void* d_ws, size_t ws_size,
                              hipStream_t stream) {
    const float* aspect    = (const float*)d_in[0];
    const float* context   = (const float*)d_in[1];
    const float* W_ct      = (const float*)d_in[2];
    const float* W_tc      = (const float*)d_in[3];
    const float* w_c1_w    = (const float*)d_in[4];
    const float* w_c1_b    = (const float*)d_in[5];
    const float* w_c2_w    = (const float*)d_in[6];
    const float* w_c2_b    = (const float*)d_in[7];
    const float* w_a1_w    = (const float*)d_in[8];
    const float* w_a1_b    = (const float*)d_in[9];
    const float* w_a2_w    = (const float*)d_in[10];
    const float* w_a2_b    = (const float*)d_in[11];
    const float* w_c       = (const float*)d_in[12];
    const float* w_a       = (const float*)d_in[13];
    const float* w_final_w = (const float*)d_in[14];
    const float* w_final_b = (const float*)d_in[15];
    float* out = (float*)d_out;

    char* ws = (char*)d_ws;
    size_t off = 0;
    auto alloc = [&](size_t bytes) -> void* {
        void* p = ws + off;
        off += (bytes + 255) & ~(size_t)255;
        return p;
    };

    __bf16* Wct_b  = (__bf16*)alloc((size_t)DD * DD * 2);
    __bf16* Wtc_b  = (__bf16*)alloc((size_t)DD * DD * 2);
    __bf16* W1cT_b = (__bf16*)alloc((size_t)DD * DD * 2);
    __bf16* W1aT_b = (__bf16*)alloc((size_t)DD * DD * 2);
    __bf16* Gb_a   = (__bf16*)alloc((size_t)NB * DD * DD * 2);
    __bf16* Gb_c   = (__bf16*)alloc((size_t)NB * DD * DD * 2);
    __bf16* Mt     = (__bf16*)alloc((size_t)NB * DD * DD * 2);
    __bf16* Nt     = (__bf16*)alloc((size_t)NB * DD * DD * 2);
    float*  Hc     = (float*)alloc((size_t)NB * LC * DD * 4);
    float*  Ha     = (float*)alloc((size_t)NB * LA * DD * 4);
    float*  Tc     = (float*)alloc((size_t)NB * LC * 4);
    float*  Ta     = (float*)alloc((size_t)NB * LA * 4);
    float*  cvec   = (float*)alloc((size_t)NB * DD * 4);
    float*  avec   = (float*)alloc((size_t)NB * DD * 4);
    float*  crep   = (float*)alloc((size_t)NB * DD * 4);
    float*  arep   = (float*)alloc((size_t)NB * DD * 4);
    (void)ws_size; (void)in_sizes; (void)n_in; (void)out_size;

    k_prep_weights<<<64, 256, 0, stream>>>(W_ct, W_tc, w_c1_w, w_a1_w,
                                           Wct_b, Wtc_b, W1cT_b, W1aT_b);

    k_gram<<<NB, 256, 0, stream>>>(aspect,  LA, Gb_a);
    k_gram<<<NB, 256, 0, stream>>>(context, LC, Gb_c);

    // Mt = G_a @ W_ct^T (= (W_ct @ G_a)^T since G symmetric); Nt = G_c @ W_tc^T
    k_mixmat<<<NB, 256, 0, stream>>>(Gb_a, Wct_b, Mt);
    k_mixmat<<<NB, 256, 0, stream>>>(Gb_c, Wtc_b, Nt);

    k_hgemm<<<dim3(LC / 128, NB), 256, 0, stream>>>(context, Mt, Hc, LC);
    k_hgemm<<<dim3(LA / 128, NB), 256, 0, stream>>>(aspect,  Nt, Ha, LA);

    k_score<<<dim3(LC / 128, NB), 256, 0, stream>>>(Hc, W1cT_b, w_c1_b, w_c2_w, w_c2_b, Tc, LC);
    k_score<<<dim3(LA / 128, NB), 256, 0, stream>>>(Ha, W1aT_b, w_a1_b, w_a2_w, w_a2_b, Ta, LA);

    k_softpool<<<NB, 256, 0, stream>>>(Tc, Hc, cvec, LC);
    k_softpool<<<NB, 256, 0, stream>>>(Ta, Ha, avec, LA);

    k_att2<<<NB, 256, 0, stream>>>(Hc, w_c, avec, crep, LC);
    k_att2<<<NB, 256, 0, stream>>>(Ha, w_a, cvec, arep, LA);

    k_final<<<NB, 256, 0, stream>>>(arep, crep, w_final_w, w_final_b, out);
}
